// EnhancedTransformerBlock_80169859548042
// MI455X (gfx1250) — matrix operand order repaired and checked by reading, not compiled, not run on hardware
//
#include <hip/hip_runtime.h>
#include <hip/hip_bf16.h>
#include <math.h>

// ---------------------------------------------------------------------------
// Types for CDNA5 WMMA (wave32, V_WMMA_F32_16X16X32_BF16)
// ---------------------------------------------------------------------------
typedef __bf16 bf16;
typedef __attribute__((ext_vector_type(16))) __bf16 bf16x16;
typedef __attribute__((ext_vector_type(8)))  float  f32x8;
typedef unsigned int       u32;
typedef unsigned long long u64;
typedef __attribute__((ext_vector_type(4))) u32 u32x4;
typedef __attribute__((ext_vector_type(8))) u32 u32x8;

#define WMMA_BF16(Af, Bf, Cf) \
  __builtin_amdgcn_wmma_f32_16x16x32_bf16(false, (Af), false, (Bf), (short)0, (Cf), false, false)

// Problem constants (match reference)
constexpr int kB  = 4;
constexpr int kS  = 1024;
constexpr int kD  = 1024;
constexpr int kH  = 16;
constexpr int kDH = 64;
constexpr int kFF = 4096;
constexpr int kE  = 8;
constexpr int kT  = kB * kS;   // 4096 tokens
constexpr int kKC = 1024;      // K-chunk staged into LDS by the TDM

union FragU { bf16x16 v; uint4 q[2]; };

// A fragment (16x32, row-major source, ld = row stride in elements).
// CDNA5 layout: lanes 0-15: M=lane, K = 0..7 (v0-3) and 16..23 (v4-7);
//               lanes 16-31: M=lane-16, K = 8..15 and 24..31.
__device__ __forceinline__ bf16x16 load_a_frag(const bf16* base, int ld) {
  const int lane = threadIdx.x & 31;
  const int row  = lane & 15;
  const int kh   = (lane >> 4) << 3;          // 0 or 8
  const bf16* p = base + (size_t)row * ld + kh;
  FragU f;
  f.q[0] = *(const uint4*)(p);                // K = kh .. kh+7
  f.q[1] = *(const uint4*)(p + 16);           // K = kh+16 .. kh+23
  return f.v;
}

// B fragment (32x16). Source is B^T stored row-major [N,K] so a column of B
// is contiguous. Layout: lanes 0-15: N=lane, K=0..7,16..23; lanes 16-31: N=lane-16, K=8..15,24..31.
__device__ __forceinline__ bf16x16 load_b_frag(const bf16* baseT, int ld) {
  const int lane = threadIdx.x & 31;
  const int col  = lane & 15;
  const int ks   = (lane >> 4) << 3;          // 0 or 8
  const bf16* p = baseT + (size_t)col * ld + ks;
  FragU f;
  f.q[0] = *(const uint4*)(p);                // K = ks .. ks+7
  f.q[1] = *(const uint4*)(p + 16);           // K = ks+16 .. ks+23
  return f.v;
}

// ---------------------------------------------------------------------------
// Tensor Data Mover: DMA a 2-D bf16 tile [nrows x rowlen] (row stride
// `stride` elements) from global memory into LDS at byte offset lds_off.
// D# layout per CDNA5 ISA (group0 + group1,
// VADDR2/3 = NULL -> 2-D form). data_size=1 => 2-byte elements.
// ---------------------------------------------------------------------------
__device__ __forceinline__ void tdm_load_2d_to_lds(const void* gptr, u32 lds_off,
                                                   u32 rowlen, u32 nrows, u32 stride) {
  const u64 ga = (u64)gptr;
  u32x4 g0;
  g0.x = 0x1u;                                        // count=1, user desc
  g0.y = lds_off;                                     // lds_addr (bytes)
  g0.z = (u32)(ga & 0xffffffffu);                     // global_addr[31:0]
  g0.w = (u32)((ga >> 32) & 0x1ffffffu) | (2u << 30); // global_addr[56:32] | type=2
  u32x8 g1;
  g1.s0 = 0x1u << 16;                                 // workgroup_mask=0, data_size=1 (2B)
  g1.s1 = (rowlen & 0xffffu) << 16;                   // tensor_dim0[15:0]
  g1.s2 = (rowlen >> 16) | ((nrows & 0xffffu) << 16); // tensor_dim0[31:16] | tensor_dim1[15:0]
  g1.s3 = (nrows >> 16) | ((rowlen & 0xffffu) << 16); // tensor_dim1[31:16] | tile_dim0
  g1.s4 = nrows & 0xffffu;                            // tile_dim1 | tile_dim2=0
  g1.s5 = stride;                                     // tensor_dim0_stride[31:0]
  g1.s6 = 0;                                          // stride[47:32] | dim1_stride lo
  g1.s7 = 0;                                          // dim1_stride hi
  asm volatile("tensor_load_to_lds %0, %1" :: "s"(g0), "s"(g1) : "memory");
}

// ---------------------------------------------------------------------------
// Weight convert + transpose: src f32 [K,N] row-major -> dst bf16 [N,K]
// blockIdx.z selects the expert (matrix) slice.
// ---------------------------------------------------------------------------
__global__ void wconv_kernel(const float* __restrict__ src, bf16* __restrict__ dst,
                             int K, int N) {
  const size_t mat = (size_t)K * N;
  const float* s = src + (size_t)blockIdx.z * mat;
  bf16* d = dst + (size_t)blockIdx.z * mat;
  for (size_t idx = (size_t)blockIdx.x * blockDim.x + threadIdx.x; idx < mat;
       idx += (size_t)gridDim.x * blockDim.x) {
    size_t n = idx / (size_t)K, k = idx % (size_t)K;
    d[idx] = (bf16)s[k * (size_t)N + n];
  }
}

// ---------------------------------------------------------------------------
// LayerNorm over D=1024, one block (256 threads) per token, bf16 out
// ---------------------------------------------------------------------------
__global__ void ln_kernel(const float* __restrict__ x, const float* __restrict__ g,
                          const float* __restrict__ b, bf16* __restrict__ o) {
  const int t = blockIdx.x, tid = threadIdx.x;
  const float* xp = x + (size_t)t * kD;
  float s = 0.f, s2 = 0.f;
  for (int d = tid; d < kD; d += 256) { float v = xp[d]; s += v; s2 += v * v; }
  __shared__ float sh[256];
  sh[tid] = s; __syncthreads();
  for (int off = 128; off > 0; off >>= 1) { if (tid < off) sh[tid] += sh[tid + off]; __syncthreads(); }
  const float mean = sh[0] * (1.f / kD);
  __syncthreads();
  sh[tid] = s2; __syncthreads();
  for (int off = 128; off > 0; off >>= 1) { if (tid < off) sh[tid] += sh[tid + off]; __syncthreads(); }
  const float var  = sh[0] * (1.f / kD) - mean * mean;
  const float rstd = rsqrtf(var + 1e-5f);
  bf16* op = o + (size_t)t * kD;
  for (int d = tid; d < kD; d += 256)
    op[d] = (bf16)((xp[d] - mean) * rstd * g[d] + b[d]);
}

// ---------------------------------------------------------------------------
// Router: logits = xn2 @ w_router + b_router ; top-2 softmax -> combine [T,E]
// One wave per token.
// ---------------------------------------------------------------------------
__global__ void router_kernel(const bf16* __restrict__ xn2, const float* __restrict__ wr,
                              const float* __restrict__ br, float* __restrict__ combine) {
  const int t = blockIdx.x;
  const int lane = threadIdx.x & 31;
  float acc[kE];
#pragma unroll
  for (int e = 0; e < kE; e++) acc[e] = 0.f;
  const bf16* xp = xn2 + (size_t)t * kD;
  for (int d = lane; d < kD; d += 32) {
    const float xv = (float)xp[d];
    const float* w = wr + (size_t)d * kE;
#pragma unroll
    for (int e = 0; e < kE; e++) acc[e] += xv * w[e];
  }
#pragma unroll
  for (int e = 0; e < kE; e++)
#pragma unroll
    for (int off = 16; off > 0; off >>= 1) acc[e] += __shfl_xor(acc[e], off);
  float logit[kE];
#pragma unroll
  for (int e = 0; e < kE; e++) logit[e] = acc[e] + br[e];
  int i1 = 0; float v1 = logit[0];
#pragma unroll
  for (int e = 1; e < kE; e++) if (logit[e] > v1) { v1 = logit[e]; i1 = e; }
  int i2 = -1; float v2 = -3.4e38f;
#pragma unroll
  for (int e = 0; e < kE; e++) if (e != i1 && logit[e] > v2) { v2 = logit[e]; i2 = e; }
  const float p1 = 1.f / (1.f + __expf(v2 - v1));
  const float p2 = 1.f - p1;
  if (lane < kE)
    combine[(size_t)t * kE + lane] = (lane == i1) ? p1 : ((lane == i2) ? p2 : 0.f);
}

// ---------------------------------------------------------------------------
// Generic WMMA GEMM: C[M,N] = A[M,K] @ B[K,N] (B supplied transposed bf16 [N,K]).
// Block = 128 threads = 4 waves. Block tile = 16 rows x 256 cols; each wave
// owns 16x64 (4 accumulators sharing one A fragment).  The 16 x kKC A slab is
// staged into LDS by the Tensor Data Mover (wave 0), synced with
// s_wait_tensorcnt + workgroup barrier.  MODE selects the epilogue.
// ---------------------------------------------------------------------------
enum { MODE_Q = 0, MODE_K = 1, MODE_V = 2, MODE_OPROJ = 3, MODE_FFN1 = 4, MODE_FFN2 = 5 };

template <int MODE>
__global__ void __launch_bounds__(128)
gemm_wmma(const bf16* __restrict__ A, const bf16* __restrict__ Bt, int Ktot, int Ncols,
          const float* __restrict__ bias, const float* __restrict__ resid,
          const float* __restrict__ combine, int expert,
          bf16* __restrict__ outb, float* __restrict__ outf) {
  const int wave = threadIdx.x >> 5;
  const int lane = threadIdx.x & 31;
  const int m0 = blockIdx.y * 16;
  const int n0 = blockIdx.x * 256 + wave * 64;

  if constexpr (MODE == MODE_FFN1 || MODE == MODE_FFN2) {
    // top-2 sparsity: skip this 16-token tile if no token routed to `expert`.
    // Uniform across the block; executed before any barrier/TDM.
    bool any = false;
    for (int t = 0; t < 16; t++)
      any |= (combine[(size_t)(m0 + t) * kE + expert] > 0.f);
    if (!any) return;
  }

  __shared__ alignas(16) bf16 sA[16 * kKC];          // 32 KB A slab
  const bf16* ap = A + (size_t)m0 * Ktot;
  const bf16* bp = Bt + (size_t)n0 * Ktot;

  f32x8 acc[4];
#pragma unroll
  for (int cg = 0; cg < 4; cg++) acc[cg] = (f32x8){0.f,0.f,0.f,0.f,0.f,0.f,0.f,0.f};

  for (int kc = 0; kc < Ktot; kc += kKC) {
    if (wave == 0) {
      tdm_load_2d_to_lds(ap + kc, (u32)(size_t)&sA[0], kKC, 16, (u32)Ktot);
      __builtin_amdgcn_s_wait_tensorcnt(0);
    }
    __syncthreads();                                  // sA ready for all waves
    for (int kk = 0; kk < kKC; kk += 32) {
      const bf16x16 af = load_a_frag(sA + kk, kKC);   // ds_load_b128 x2
      const bf16* bq = bp + kc + kk;
      if (kk + 32 < kKC)
        __builtin_prefetch((const void*)(bq + 32 + (size_t)(lane & 15) * Ktot), 0, 1);
      acc[0] = WMMA_BF16(af, load_b_frag(bq, Ktot), acc[0]);
      acc[1] = WMMA_BF16(af, load_b_frag(bq + (size_t)16 * Ktot, Ktot), acc[1]);
      acc[2] = WMMA_BF16(af, load_b_frag(bq + (size_t)32 * Ktot, Ktot), acc[2]);
      acc[3] = WMMA_BF16(af, load_b_frag(bq + (size_t)48 * Ktot, Ktot), acc[3]);
    }
    __syncthreads();                                  // before TDM overwrites sA
  }

  const int nlo = lane & 15;
  const int hi8 = (lane >> 4) << 3;
#pragma unroll
  for (int cg = 0; cg < 4; cg++) {
#pragma unroll
    for (int i = 0; i < 8; i++) {
      const int row = m0 + i + hi8;            // token index
      const int col = n0 + cg * 16 + nlo;      // output feature
      float v = acc[cg][i] + (bias ? bias[col] : 0.f);
      if constexpr (MODE == MODE_Q || MODE == MODE_K) {
        const int b = row >> 10, s = row & 1023, h = col >> 6, dh = col & 63;
        outb[(((size_t)(b * kH + h) * kS) + s) * kDH + dh] = (bf16)v;
      } else if constexpr (MODE == MODE_V) {   // store V transposed: [B,H,DH,S]
        const int b = row >> 10, s = row & 1023, h = col >> 6, dh = col & 63;
        outb[(((size_t)(b * kH + h) * kDH) + dh) * kS + s] = (bf16)v;
      } else if constexpr (MODE == MODE_OPROJ) {
        outf[(size_t)row * Ncols + col] = v + resid[(size_t)row * Ncols + col];
      } else if constexpr (MODE == MODE_FFN1) { // exact GELU
        const float gl = 0.5f * v * (1.f + erff(v * 0.70710678118654752f));
        outb[(size_t)row * Ncols + col] = (bf16)gl;
      } else {                                  // MODE_FFN2: weighted accumulate
        const float c = combine[(size_t)row * kE + expert];
        outf[(size_t)row * Ncols + col] += c * v;
      }
    }
  }
}

// ---------------------------------------------------------------------------
// Flash attention: one wave per 16 query rows of one (b,h).
// scores = Q @ K^T * 1/8 (+mask), online softmax, out = P @ V.
// P re-staged through LDS to convert C-layout -> A-fragment layout.
// ---------------------------------------------------------------------------
__global__ void __launch_bounds__(32)
attn_kernel(const bf16* __restrict__ q, const bf16* __restrict__ kmat,
            const bf16* __restrict__ vt, const int* __restrict__ mask,
            bf16* __restrict__ ao) {
  const int qt = blockIdx.x, h = blockIdx.y, b = blockIdx.z;
  const int lane = threadIdx.x & 31;
  const int nlo = lane & 15;
  const int hi8 = (lane >> 4) << 3;

  const bf16* qp  = q    + (((size_t)(b * kH + h)) * kS + (size_t)qt * 16) * kDH;
  const bf16* kp  = kmat + ((size_t)(b * kH + h)) * kS * kDH;
  const bf16* vtp = vt   + ((size_t)(b * kH + h)) * kDH * kS;
  const int*  mp  = mask + (size_t)b * kS;

  const bf16x16 qa0 = load_a_frag(qp, kDH);        // dh 0..31
  const bf16x16 qa1 = load_a_frag(qp + 32, kDH);   // dh 32..63

  f32x8 o0 = {0.f,0.f,0.f,0.f,0.f,0.f,0.f,0.f};
  f32x8 o1 = o0, o2 = o0, o3 = o0;
  float mrow[8], lrow[8];
#pragma unroll
  for (int i = 0; i < 8; i++) { mrow[i] = -1e30f; lrow[i] = 0.f; }

  __shared__ alignas(16) bf16 sP[16][32];

  for (int j = 0; j < kS; j += 32) {
    f32x8 s0 = {0.f,0.f,0.f,0.f,0.f,0.f,0.f,0.f};
    f32x8 s1 = s0;
    // keys j..j+15 and j+16..j+31 ; contraction over dh (2x32)
    s0 = WMMA_BF16(qa0, load_b_frag(kp + (size_t)j * kDH, kDH), s0);
    s0 = WMMA_BF16(qa1, load_b_frag(kp + (size_t)j * kDH + 32, kDH), s0);
    s1 = WMMA_BF16(qa0, load_b_frag(kp + (size_t)(j + 16) * kDH, kDH), s1);
    s1 = WMMA_BF16(qa1, load_b_frag(kp + (size_t)(j + 16) * kDH + 32, kDH), s1);

    const float msk0 = mp[j + nlo]      ? 0.f : -1e30f;
    const float msk1 = mp[j + 16 + nlo] ? 0.f : -1e30f;

#pragma unroll
    for (int i = 0; i < 8; i++) {
      float a = s0[i] * 0.125f + msk0;
      float c = s1[i] * 0.125f + msk1;
      float mx = fmaxf(a, c);
#pragma unroll
      for (int d = 1; d < 16; d <<= 1) mx = fmaxf(mx, __shfl_xor(mx, d));
      const float mn   = fmaxf(mrow[i], mx);
      const float corr = __expf(mrow[i] - mn);
      const float ea = __expf(a - mn), eb = __expf(c - mn);
      float rs = ea + eb;
#pragma unroll
      for (int d = 1; d < 16; d <<= 1) rs += __shfl_xor(rs, d);
      lrow[i] = lrow[i] * corr + rs;
      mrow[i] = mn;
      o0[i] *= corr; o1[i] *= corr; o2[i] *= corr; o3[i] *= corr;
      sP[i + hi8][nlo]      = (bf16)ea;
      sP[i + hi8][nlo + 16] = (bf16)eb;
    }
    __syncthreads();
    const bf16x16 pa = load_a_frag(&sP[0][0], 32);   // P tile 16x32 (K = key)
    o0 = WMMA_BF16(pa, load_b_frag(vtp + j, kS), o0);             // dh 0..15
    o1 = WMMA_BF16(pa, load_b_frag(vtp + 16 * kS + j, kS), o1);   // dh 16..31
    o2 = WMMA_BF16(pa, load_b_frag(vtp + 32 * kS + j, kS), o2);   // dh 32..47
    o3 = WMMA_BF16(pa, load_b_frag(vtp + 48 * kS + j, kS), o3);   // dh 48..63
    __syncthreads();
  }

  const size_t base = ((size_t)b * kS + (size_t)qt * 16) * kD + (size_t)h * kDH;
#pragma unroll
  for (int i = 0; i < 8; i++) {
    const int r = i + hi8;
    const float inv = 1.f / lrow[i];
    ao[base + (size_t)r * kD + nlo]      = (bf16)(o0[i] * inv);
    ao[base + (size_t)r * kD + 16 + nlo] = (bf16)(o1[i] * inv);
    ao[base + (size_t)r * kD + 32 + nlo] = (bf16)(o2[i] * inv);
    ao[base + (size_t)r * kD + 48 + nlo] = (bf16)(o3[i] * inv);
  }
}

// ---------------------------------------------------------------------------
// Host-side orchestration (all on `stream`; workspace ~216 MB)
// ---------------------------------------------------------------------------
extern "C" void kernel_launch(void* const* d_in, const int* in_sizes, int n_in,
                              void* d_out, int out_size, void* d_ws, size_t ws_size,
                              hipStream_t stream) {
  (void)in_sizes; (void)n_in; (void)out_size; (void)ws_size;
  const float* x    = (const float*)d_in[0];
  const int*   mask = (const int*)  d_in[1];
  const float* ln1g = (const float*)d_in[2];
  const float* ln1b = (const float*)d_in[3];
  const float* ln2g = (const float*)d_in[4];
  const float* ln2b = (const float*)d_in[5];
  const float* wq   = (const float*)d_in[6];
  const float* bq   = (const float*)d_in[7];
  const float* wk   = (const float*)d_in[8];
  const float* bk   = (const float*)d_in[9];
  const float* wv   = (const float*)d_in[10];
  const float* bv   = (const float*)d_in[11];
  const float* wo   = (const float*)d_in[12];
  const float* bo   = (const float*)d_in[13];
  const float* wr   = (const float*)d_in[14];
  const float* br   = (const float*)d_in[15];
  const float* w1   = (const float*)d_in[16];
  const float* b1   = (const float*)d_in[17];
  const float* w2   = (const float*)d_in[18];
  const float* b2   = (const float*)d_in[19];
  float* out = (float*)d_out;

  char* p = (char*)d_ws;
  auto alloc = [&](size_t bytes) -> char* {
    char* r = p; p += (bytes + 255) & ~(size_t)255; return r;
  };
  bf16* WQT  = (bf16*)alloc((size_t)kD * kD * 2);
  bf16* WKT  = (bf16*)alloc((size_t)kD * kD * 2);
  bf16* WVT  = (bf16*)alloc((size_t)kD * kD * 2);
  bf16* WOT  = (bf16*)alloc((size_t)kD * kD * 2);
  bf16* W1T  = (bf16*)alloc((size_t)kE * kFF * kD * 2);   // [E][FF][D]
  bf16* W2T  = (bf16*)alloc((size_t)kE * kD * kFF * 2);   // [E][D][FF]
  bf16* XN   = (bf16*)alloc((size_t)kT * kD * 2);
  bf16* Qb   = (bf16*)alloc((size_t)kT * kD * 2);         // [B,H,S,DH]
  bf16* Kb   = (bf16*)alloc((size_t)kT * kD * 2);         // [B,H,S,DH]
  bf16* VTb  = (bf16*)alloc((size_t)kT * kD * 2);         // [B,H,DH,S]
  bf16* AOb  = (bf16*)alloc((size_t)kT * kD * 2);         // [B,S,D]
  bf16* XN2  = (bf16*)alloc((size_t)kT * kD * 2);
  bf16* Hb   = (bf16*)alloc((size_t)kT * kFF * 2);        // expert hidden (reused)
  float* CMB = (float*)alloc((size_t)kT * kE * 4);

  // 1) weight convert + transpose to bf16 [N,K]
  wconv_kernel<<<dim3(4096, 1, 1), 256, 0, stream>>>(wq, WQT, kD, kD);
  wconv_kernel<<<dim3(4096, 1, 1), 256, 0, stream>>>(wk, WKT, kD, kD);
  wconv_kernel<<<dim3(4096, 1, 1), 256, 0, stream>>>(wv, WVT, kD, kD);
  wconv_kernel<<<dim3(4096, 1, 1), 256, 0, stream>>>(wo, WOT, kD, kD);
  wconv_kernel<<<dim3(16384, 1, kE), 256, 0, stream>>>(w1, W1T, kD, kFF);
  wconv_kernel<<<dim3(16384, 1, kE), 256, 0, stream>>>(w2, W2T, kFF, kD);

  // 2) LN1
  ln_kernel<<<kT, 256, 0, stream>>>(x, ln1g, ln1b, XN);

  // 3) QKV projections (WMMA bf16, TDM-staged A)
  const dim3 gD(kD / 256, kT / 16);    // (4, 256)
  gemm_wmma<MODE_Q><<<gD, 128, 0, stream>>>(XN, WQT, kD, kD, bq, nullptr, nullptr, 0, Qb, nullptr);
  gemm_wmma<MODE_K><<<gD, 128, 0, stream>>>(XN, WKT, kD, kD, bk, nullptr, nullptr, 0, Kb, nullptr);
  gemm_wmma<MODE_V><<<gD, 128, 0, stream>>>(XN, WVT, kD, kD, bv, nullptr, nullptr, 0, VTb, nullptr);

  // 4) attention (flash, WMMA)
  attn_kernel<<<dim3(kS / 16, kH, kB), 32, 0, stream>>>(Qb, Kb, VTb, mask, AOb);

  // 5) output projection + residual -> x1 lives in d_out (f32)
  gemm_wmma<MODE_OPROJ><<<gD, 128, 0, stream>>>(AOb, WOT, kD, kD, bo, x, nullptr, 0, nullptr, out);

  // 6) LN2 + router -> combine weights
  ln_kernel<<<kT, 256, 0, stream>>>(out, ln2g, ln2b, XN2);
  router_kernel<<<kT, 32, 0, stream>>>(XN2, wr, br, CMB);

  // 7) MoE: per expert, FFN1 (gelu) then FFN2 (combine-weighted accumulate into d_out)
  const dim3 gFF(kFF / 256, kT / 16);  // (16, 256)
  for (int e = 0; e < kE; e++) {
    gemm_wmma<MODE_FFN1><<<gFF, 128, 0, stream>>>(
        XN2, W1T + (size_t)e * kFF * kD, kD, kFF, b1 + (size_t)e * kFF,
        nullptr, CMB, e, Hb, nullptr);
    gemm_wmma<MODE_FFN2><<<gD, 128, 0, stream>>>(
        Hb, W2T + (size_t)e * kD * kFF, kFF, kD, b2 + (size_t)e * kD,
        nullptr, CMB, e, nullptr, out);
  }
}
